// Glm4MoE_62637803045576
// MI455X (gfx1250) — compile-verified, matrix operand order checked by reading, not run on hardware
//
#include <hip/hip_runtime.h>
#include <hip/hip_bf16.h>

#define T_    1024
#define H_    2048
#define E_    32
#define I_    768
#define TOPK_ 8
#define LDGU_ (2 * I_)

typedef __attribute__((ext_vector_type(16))) __bf16 v16bf;
typedef __attribute__((ext_vector_type(8)))  __bf16 v8bf;
typedef __attribute__((ext_vector_type(4)))  __bf16 v4bf;
typedef __attribute__((ext_vector_type(8)))  float  v8f;
typedef __attribute__((ext_vector_type(4)))  unsigned int u32x4;
typedef __attribute__((ext_vector_type(8)))  int i32x8;
typedef __attribute__((ext_vector_type(4)))  int i32x4;

#if __has_builtin(__builtin_amdgcn_tensor_load_to_lds)
#define HAVE_TDM 1
#else
#define HAVE_TDM 0
#endif

// Load a 16-wide bf16 WMMA fragment from an LDS row laid out K-contiguous.
// ISA 7.12.2: half-wave hv=0 holds K=0..7 & 16..23, hv=1 holds K=8..15 & 24..31.
static __device__ inline v16bf ldfrag(const __bf16* rowbase, int hv) {
    v8bf lo = *(const v8bf*)(rowbase + 8 * hv);
    v8bf hi = *(const v8bf*)(rowbase + 16 + 8 * hv);
    return __builtin_shufflevector(lo, hi, 0, 1, 2, 3, 4, 5, 6, 7,
                                           8, 9, 10, 11, 12, 13, 14, 15);
}

static __device__ inline v8f wmma_bf16(v16bf a, v16bf b, v8f c) {
    return __builtin_amdgcn_wmma_f32_16x16x32_bf16(
        false, a, false, b, (short)0, c, false, false);
}

static __device__ inline v4bf cvt4(float a, float b, float c, float d) {
    v4bf v;
    v[0] = (__bf16)a; v[1] = (__bf16)b; v[2] = (__bf16)c; v[3] = (__bf16)d;
    return v;
}

static __device__ inline float fast_sigmoid(float x) {
    return __builtin_amdgcn_rcpf(1.f + __expf(-x));   // v_rcp_f32, no IEEE div
}

// ---------------------------------------------------------------------------
// Kernel 1: fp32 gate logits + grouped sigmoid top-k + per-expert compaction
// ---------------------------------------------------------------------------
__global__ void glm_gate_topk(const float* __restrict__ x,
                              const float* __restrict__ gw,
                              const float* __restrict__ gbias,
                              int* __restrict__ cnt,
                              int* __restrict__ elist,
                              float* __restrict__ ewt) {
    const int t   = blockIdx.x;
    const int tid = threadIdx.x;          // 256 threads
    const int e   = tid & 31;
    const int seg = tid >> 5;             // 8 H-segments of 256
    const float4* xr = (const float4*)(x  + (size_t)t * H_ + seg * 256);
    const float4* wr = (const float4*)(gw + (size_t)e * H_ + seg * 256);
    float p = 0.f;
#pragma unroll 8
    for (int h = 0; h < 64; ++h) {
        float4 a = xr[h], b = wr[h];
        p += a.x * b.x + a.y * b.y + a.z * b.z + a.w * b.w;
    }

    __shared__ float part[256];
    __shared__ float score[E_];
    part[tid] = p;
    __syncthreads();
    if (tid < E_) {
        float s = part[tid];
        for (int j = 1; j < 8; ++j) s += part[tid + 32 * j];
        score[tid] = fast_sigmoid(s);
    }
    __syncthreads();
    if (tid == 0) {
        float sc[E_], w[TOPK_];
        int   ids[TOPK_];
        for (int i = 0; i < E_; ++i) sc[i] = score[i] + gbias[i];
        float gs[4];
        for (int g = 0; g < 4; ++g) {
            float m1 = -1e30f, m2 = -1e30f;
            for (int j = 0; j < 8; ++j) {
                float v = sc[g * 8 + j];
                if (v > m1) { m2 = m1; m1 = v; }
                else if (v > m2) m2 = v;
            }
            gs[g] = m1 + m2;
        }
        int g1 = 0;
        for (int g = 1; g < 4; ++g) if (gs[g] > gs[g1]) g1 = g;
        gs[g1] = -1e30f;
        int g2 = 0;
        for (int g = 1; g < 4; ++g) if (gs[g] > gs[g2]) g2 = g;
        float cand[E_];
        for (int i = 0; i < E_; ++i) {
            int g = i >> 3;
            cand[i] = (g == g1 || g == g2) ? sc[i] : -1e30f;
        }
        float wsum = 0.f;
        for (int k = 0; k < TOPK_; ++k) {
            int am = 0;
            for (int i = 1; i < E_; ++i) if (cand[i] > cand[am]) am = i;
            ids[k] = am;
            w[k]   = score[am];
            cand[am] = -1e30f;
            wsum += w[k];
        }
        float inv = 1.f / wsum;
        for (int k = 0; k < TOPK_; ++k) {
            int ee  = ids[k];
            int pos = atomicAdd(&cnt[ee], 1);
            elist[(size_t)ee * T_ + pos] = t * TOPK_ + k;   // token*8 + slot
            ewt[(size_t)ee * T_ + pos]   = w[k] * inv;
        }
    }
}

// ---------------------------------------------------------------------------
// Gate/Up GEMM tile (64 rows x 64 i-cols), K = H_, double-buffered LDS,
// fused SiLU*mul -> bf16 activations
// ---------------------------------------------------------------------------
__device__ void moe_gateup_tile(const float* __restrict__ x,
                                const float* __restrict__ w,     // [H_][2*I_]
                                const int* __restrict__ elist,   // nullptr => identity
                                int nrows,
                                __bf16* __restrict__ act) {      // [cap][I_]
    const int m0 = blockIdx.y * 64;
    if (m0 >= nrows) return;
    const int n0 = blockIdx.x * 64;

    __shared__ __align__(16) __bf16 As[2][64][32];
    __shared__ __align__(16) __bf16 Bg[2][64][32];   // [n][k]
    __shared__ __align__(16) __bf16 Bu[2][64][32];   // [n][k]
    __shared__ int rowtok[64];

    const int tid = threadIdx.x;
    if (tid < 64) {
        int r = m0 + tid;
        rowtok[tid] = (r < nrows) ? (elist ? (elist[r] >> 3) : r) : -1;
    }
    __syncthreads();

    // loop-invariant staging geometry
    const int ra  = tid >> 3;            // A rows ra and ra+32
    const int kqa = (tid & 7) * 4;       // k-quad within A row
    const int tk0 = rowtok[ra], tk1 = rowtok[ra + 32];
    const float* aptr0 = x + (size_t)(tk0 < 0 ? 0 : tk0) * H_ + kqa;
    const float* aptr1 = x + (size_t)(tk1 < 0 ? 0 : tk1) * H_ + kqa;
    const float sc0 = tk0 < 0 ? 0.f : 1.f;
    const float sc1 = tk1 < 0 ? 0.f : 1.f;
    const int nb = tid & 63;             // B column
    const int kb = (tid >> 6) * 4;       // B k-quad base (0,4,8,12), +16 on pass 1
    const float* bptr = w + (size_t)(n0 + nb);

    auto stage = [&](int buf, int k0) {
        float4 a0 = *(const float4*)(aptr0 + k0);
        float4 a1 = *(const float4*)(aptr1 + k0);
        float g[8], u[8];
#pragma unroll
        for (int p = 0; p < 2; ++p) {
#pragma unroll
            for (int j = 0; j < 4; ++j) {
                const float* wp = bptr + (size_t)(k0 + kb + p * 16 + j) * LDGU_;
                g[p * 4 + j] = wp[0];     // coalesced across lanes (consecutive n)
                u[p * 4 + j] = wp[I_];
            }
        }
        *(v4bf*)&As[buf][ra][kqa]      = cvt4(a0.x * sc0, a0.y * sc0, a0.z * sc0, a0.w * sc0);
        *(v4bf*)&As[buf][ra + 32][kqa] = cvt4(a1.x * sc1, a1.y * sc1, a1.z * sc1, a1.w * sc1);
#pragma unroll
        for (int p = 0; p < 2; ++p) {
            int kk = kb + p * 16;
            *(v4bf*)&Bg[buf][nb][kk] = cvt4(g[p*4], g[p*4+1], g[p*4+2], g[p*4+3]);
            *(v4bf*)&Bu[buf][nb][kk] = cvt4(u[p*4], u[p*4+1], u[p*4+2], u[p*4+3]);
        }
    };

    const int lane = tid & 31, wave = tid >> 5;
    const int wm = wave & 3, wn = wave >> 2;
    const int hv = lane >> 4, ln = lane & 15;
    v8f cg0 = {}, cg1 = {}, cu0 = {}, cu1 = {};

    stage(0, 0);
    __syncthreads();
    constexpr int NT = H_ / 32;
    for (int kt = 0; kt < NT; ++kt) {
        const int buf = kt & 1;
        if (kt + 1 < NT) stage(buf ^ 1, (kt + 1) * 32);   // prefetch next tile
        v16bf a  = ldfrag(&As[buf][wm * 16 + ln][0], hv);
        v16bf b0 = ldfrag(&Bg[buf][wn * 32 + ln][0], hv);
        v16bf b1 = ldfrag(&Bg[buf][wn * 32 + 16 + ln][0], hv);
        v16bf b2 = ldfrag(&Bu[buf][wn * 32 + ln][0], hv);
        v16bf b3 = ldfrag(&Bu[buf][wn * 32 + 16 + ln][0], hv);
        cg0 = wmma_bf16(a, b0, cg0);
        cg1 = wmma_bf16(a, b1, cg1);
        cu0 = wmma_bf16(a, b2, cu0);
        cu1 = wmma_bf16(a, b3, cu1);
        __syncthreads();
    }

#pragma unroll
    for (int f = 0; f < 2; ++f) {                // act = silu(g) * u
        v8f g = f ? cg1 : cg0;
        v8f u = f ? cu1 : cu0;
        int col = n0 + wn * 32 + f * 16 + ln;
#pragma unroll
        for (int r = 0; r < 8; ++r) {
            int row = m0 + wm * 16 + 8 * hv + r;
            if (row < nrows) {
                float gv = g[r];
                act[(size_t)row * I_ + col] = (__bf16)(gv * fast_sigmoid(gv) * u[r]);
            }
        }
    }
}

// ---------------------------------------------------------------------------
// Down GEMM tile (64 rows x 64 h-cols), K = I_, double-buffered LDS.
// A tile (bf16 in global) is DMA'd by the Tensor Data Mover straight into
// LDS (TENSORcnt), overlapping the WMMAs; B tile converts fp32->bf16 in VGPRs.
// ---------------------------------------------------------------------------
__device__ void moe_down_tile(const __bf16* __restrict__ act,   // expert base
                              const float* __restrict__ w,      // [I_][H_]
                              const int* __restrict__ elist,    // nullptr => identity
                              const float* __restrict__ ewt,    // nullptr => 1.0
                              int nrows,
                              float* __restrict__ yout) {
    const int m0 = blockIdx.y * 64;
    if (m0 >= nrows) return;
    const int n0 = blockIdx.x * 64;

    __shared__ __align__(16) __bf16 As[2][64][32];
    __shared__ __align__(16) __bf16 Bs[2][64][32];   // [n][k]
    __shared__ int   rowmap[64];
    __shared__ float rowscl[64];

    const int tid = threadIdx.x;
    if (tid < 64) {
        int r = m0 + tid;
        bool ok = r < nrows;
        rowmap[tid] = ok ? (elist ? elist[r] : r) : 0;
        rowscl[tid] = ok ? (2.5f * (ewt ? ewt[r] : 0.4f)) : 0.f;  // 2.5*w, shared: 1.0
    }
    __syncthreads();

    const int lane = tid & 31, wave = tid >> 5;
    const int nb = tid & 63;
    const int kb = (tid >> 6) * 4;
    const float* bptr = w + (size_t)(n0 + nb);

#if HAVE_TDM
    // Tensor DMA descriptor (ISA ch.8): 64x32 bf16 tile of act -> LDS As[buf].
    const unsigned int lds_base = (unsigned int)(unsigned long long)&As[0][0][0];
    auto stage_a_tdm = [&](int buf, int k0) {
        if (wave == 0) {   // TDM is a per-wave DMA op; one wave issues it
            unsigned long long ga =
                (unsigned long long)(const void*)(act + (size_t)m0 * I_ + k0);
            u32x4 g0;
            g0[0] = 1u;                                  // count=1, user mode
            g0[1] = lds_base + (unsigned)buf * (64u * 32u * 2u);
            g0[2] = (unsigned int)ga;                    // global_addr[31:0]
            g0[3] = ((unsigned int)(ga >> 32) & 0x01FFFFFFu) | 0x80000000u; // type=2
            i32x8 g1;
            g1[0] = 0x00010000;          // workgroup_mask=0, data_size=1 (2 bytes)
            g1[1] = (int)((unsigned)I_ << 16);           // tensor_dim0 = I_ (lo16)
            g1[2] = (int)(64u << 16);                    // tensor_dim1 = 64 rows
            g1[3] = (int)(32u << 16);                    // tile_dim0 = 32
            g1[4] = 64;                                  // tile_dim1 = 64
            g1[5] = I_;                                  // tensor_dim0_stride = I_
            g1[6] = 0;
            g1[7] = 0;
            i32x4 gz4 = {};
            i32x8 gz8 = {};
            __builtin_amdgcn_tensor_load_to_lds(g0, g1, gz4, gz4, gz8, 0);
        }
    };
#else
    const int ra  = tid >> 3;
    const int kqa = (tid & 7) * 4;
    const bool av0 = (m0 + ra) < nrows;
    const bool av1 = (m0 + ra + 32) < nrows;
    const __bf16* aptr0 = act + (size_t)(av0 ? m0 + ra      : 0) * I_ + kqa;
    const __bf16* aptr1 = act + (size_t)(av1 ? m0 + ra + 32 : 0) * I_ + kqa;
    auto stage_a_vgpr = [&](int buf, int k0) {
        v4bf a0 = *(const v4bf*)(aptr0 + k0);
        v4bf a1 = *(const v4bf*)(aptr1 + k0);
        v4bf z = {};
        if (!av0) a0 = z;
        if (!av1) a1 = z;
        *(v4bf*)&As[buf][ra][kqa]      = a0;
        *(v4bf*)&As[buf][ra + 32][kqa] = a1;
    };
#endif

    auto stage_b = [&](int buf, int k0) {
        float b[8];
#pragma unroll
        for (int p = 0; p < 2; ++p) {
#pragma unroll
            for (int j = 0; j < 4; ++j)
                b[p * 4 + j] = bptr[(size_t)(k0 + kb + p * 16 + j) * H_];
        }
#pragma unroll
        for (int p = 0; p < 2; ++p) {
            int kk = kb + p * 16;
            *(v4bf*)&Bs[buf][nb][kk] = cvt4(b[p*4], b[p*4+1], b[p*4+2], b[p*4+3]);
        }
    };

    const int wm = wave & 3, wn = wave >> 2;
    const int hv = lane >> 4, ln = lane & 15;
    v8f c0 = {}, c1 = {};

#if HAVE_TDM
    stage_a_tdm(0, 0);
    stage_b(0, 0);
    if (wave == 0) __builtin_amdgcn_s_wait_tensorcnt(0);
    __syncthreads();
#else
    stage_a_vgpr(0, 0);
    stage_b(0, 0);
    __syncthreads();
#endif

    constexpr int NT = I_ / 32;
    for (int kt = 0; kt < NT; ++kt) {
        const int buf = kt & 1;
        if (kt + 1 < NT) {
#if HAVE_TDM
            stage_a_tdm(buf ^ 1, (kt + 1) * 32);   // DMA overlaps the WMMAs below
#else
            stage_a_vgpr(buf ^ 1, (kt + 1) * 32);
#endif
            stage_b(buf ^ 1, (kt + 1) * 32);
        }
        v16bf a  = ldfrag(&As[buf][wm * 16 + ln][0], hv);
        v16bf b0 = ldfrag(&Bs[buf][wn * 32 + ln][0], hv);
        v16bf b1 = ldfrag(&Bs[buf][wn * 32 + 16 + ln][0], hv);
        c0 = wmma_bf16(a, b0, c0);
        c1 = wmma_bf16(a, b1, c1);
#if HAVE_TDM
        if (wave == 0) __builtin_amdgcn_s_wait_tensorcnt(0);
#endif
        __syncthreads();
    }

#pragma unroll
    for (int f = 0; f < 2; ++f) {
        v8f c = f ? c1 : c0;
        int col = n0 + wn * 32 + f * 16 + ln;
#pragma unroll
        for (int r = 0; r < 8; ++r) {
            int local = wm * 16 + 8 * hv + r;
            if (m0 + local < nrows)
                yout[(size_t)rowmap[local] * H_ + col] = rowscl[local] * c[r];
        }
    }
}

// ---------------------------------------------------------------------------
// Wrappers
// ---------------------------------------------------------------------------
__global__ void glm_gateup_routed(const float* x, const float* w_gu,
                                  const int* elist, const int* cnt,
                                  unsigned short* act) {
    int e = blockIdx.z;
    moe_gateup_tile(x, w_gu + (size_t)e * H_ * LDGU_, elist + (size_t)e * T_,
                    cnt[e], (__bf16*)act + (size_t)e * T_ * I_);
}
__global__ void glm_gateup_shared(const float* x, const float* s_gu,
                                  unsigned short* act) {
    moe_gateup_tile(x, s_gu, nullptr, T_, (__bf16*)act);
}
__global__ void glm_down_routed(const unsigned short* act, const float* w_dn,
                                const int* elist, const float* ewt,
                                const int* cnt, float* y_r) {
    int e = blockIdx.z;
    moe_down_tile((const __bf16*)act + (size_t)e * T_ * I_,
                  w_dn + (size_t)e * I_ * H_,
                  elist + (size_t)e * T_, ewt + (size_t)e * T_, cnt[e], y_r);
}
__global__ void glm_down_shared(const unsigned short* act, const float* s_dn,
                                float* y_s) {
    moe_down_tile((const __bf16*)act, s_dn, nullptr, nullptr, T_, y_s);
}

// out[t][h] = sum_slot y_r[t*8+slot][h] (already scaled by 2.5*w) + y_s[t][h]
__global__ void glm_combine(const float* __restrict__ y_r,
                            const float* __restrict__ y_s,
                            float* __restrict__ out) {
    size_t i = ((size_t)blockIdx.x * blockDim.x + threadIdx.x) * 4;
    int t = (int)(i >> 11);          // / H_
    int h = (int)(i & (H_ - 1));
    float4 acc = *(const float4*)(y_s + i);
    const float* p = y_r + ((size_t)t * TOPK_) * H_ + h;
#pragma unroll
    for (int s = 0; s < TOPK_; ++s) {
        float4 v = *(const float4*)(p + (size_t)s * H_);
        acc.x += v.x; acc.y += v.y; acc.z += v.z; acc.w += v.w;
    }
    *(float4*)(out + i) = acc;
}

// ---------------------------------------------------------------------------
extern "C" void kernel_launch(void* const* d_in, const int* in_sizes, int n_in,
                              void* d_out, int out_size, void* d_ws, size_t ws_size,
                              hipStream_t stream) {
    const float* x     = (const float*)d_in[0];
    const float* gw    = (const float*)d_in[1];
    const float* gbias = (const float*)d_in[2];
    const float* w_gu  = (const float*)d_in[3];
    const float* w_dn  = (const float*)d_in[4];
    const float* s_gu  = (const float*)d_in[5];
    const float* s_dn  = (const float*)d_in[6];
    float* out = (float*)d_out;

    char* p = (char*)d_ws;
    int*   cnt   = (int*)p;                       p += 256;
    int*   elist = (int*)p;                       p += (size_t)E_ * T_ * sizeof(int);
    float* ewt   = (float*)p;                     p += (size_t)E_ * T_ * sizeof(float);
    unsigned short* act_r = (unsigned short*)p;   p += (size_t)E_ * T_ * I_ * 2;
    unsigned short* act_s = (unsigned short*)p;   p += (size_t)T_ * I_ * 2;
    float* y_r = (float*)p;                       p += (size_t)T_ * TOPK_ * H_ * sizeof(float);
    float* y_s = (float*)p;

    (void)hipMemsetAsync(cnt, 0, E_ * sizeof(int), stream);
    glm_gate_topk<<<T_, 256, 0, stream>>>(x, gw, gbias, cnt, elist, ewt);
    glm_gateup_shared<<<dim3(I_ / 64, T_ / 64, 1), 256, 0, stream>>>(x, s_gu, act_s);
    glm_gateup_routed<<<dim3(I_ / 64, T_ / 64, E_), 256, 0, stream>>>(x, w_gu, elist, cnt, act_r);
    glm_down_shared<<<dim3(H_ / 64, T_ / 64, 1), 256, 0, stream>>>(act_s, s_dn, y_s);
    glm_down_routed<<<dim3(H_ / 64, T_ / 64, E_), 256, 0, stream>>>(act_r, w_dn, elist, ewt, cnt, y_r);
    glm_combine<<<(T_ * H_ / 4) / 256, 256, 0, stream>>>(y_r, y_s, out);
}
